// Zamba2MambaDecoderLayer_29712583754021
// MI455X (gfx1250) — matrix operand order repaired and checked by reading, not compiled, not run on hardware
//
#include <hip/hip_runtime.h>
#include <hip/hip_bf16.h>

// ---------------------------------------------------------------------------
// Zamba2 Mamba decoder layer, MI455X (gfx1250), wave32 + WMMA bf16 path.
// All GEMM-like contractions run on v_wmma_f32_16x16x32_bf16.
// GEMMs use 64x64 wave tiles (16 WMMA per k-step) + global_prefetch_b8.
// ---------------------------------------------------------------------------

typedef __bf16 bf16_t;
typedef __attribute__((ext_vector_type(16))) __bf16 v16bf;
typedef __attribute__((ext_vector_type(8)))  __bf16 v8bf;
typedef __attribute__((ext_vector_type(8)))  float  v8f;

#define Bn     2
#define Sn     2048
#define Hn     2048
#define In     4096
#define NHn    64
#define HDn    64
#define Nn     64
#define Kn     4
#define CHUNKn 256
#define NCn    8            // chunks per batch (S/CHUNK)
#define PROJn  8384
#define CONVDn 4224
#define ROWSn  4096         // B*S
#define EPSf   1e-5f

// ---------------------------------------------------------------------------
// WMMA helpers (CDNA5 fragment layouts, ISA 7.12.2)
// ---------------------------------------------------------------------------
__device__ __forceinline__ v8f wmma_bf16(v16bf a, v16bf b, v8f c) {
  return __builtin_amdgcn_wmma_f32_16x16x32_bf16(false, a, false, b, (short)0, c,
                                                 false, false);
}

// A fragment: 16x32 (MxK), row-major source with leading dim ld.
// lanes 0-15: K 0..15; lanes 16-31: K 16..31.
__device__ __forceinline__ v16bf load_frag_a(const bf16_t* __restrict__ A, int ld,
                                             int row, int k, int lane) {
  int r   = row + (lane & 15);
  int off = (lane & 16) ? 16 : 0;
  const bf16_t* p = A + (size_t)r * ld + k + off;
  union { v16bf v; v8bf h[2]; } u;
  u.h[0] = *(const v8bf*)(p);
  u.h[1] = *(const v8bf*)(p + 8);
  return u.v;
}

// B fragment: 32x16 (KxN) sourced from an NxK row-major array (Bt):
// lane n holds column n; lanes 0-15 K 0..15, lanes 16-31 K 16..31.
__device__ __forceinline__ v16bf load_frag_bt(const bf16_t* __restrict__ Bt, int ld,
                                              int col, int k, int lane) {
  int c    = col + (lane & 15);
  int koff = k + ((lane & 16) ? 16 : 0);
  return *(const v16bf*)(Bt + (size_t)c * ld + koff);
}

// ---------------------------------------------------------------------------
// elementwise f32 -> bf16
// ---------------------------------------------------------------------------
__global__ void k_f32_to_bf16(const float* __restrict__ src,
                              bf16_t* __restrict__ dst, size_t n) {
  size_t i = (size_t)blockIdx.x * blockDim.x + threadIdx.x;
  size_t stride = (size_t)gridDim.x * blockDim.x;
  for (; i < n; i += stride) dst[i] = (bf16_t)src[i];
}

// ---------------------------------------------------------------------------
// hs = rmsnorm(hidden + transformer_hidden) * rms_w  -> bf16   (row = 2048)
// ---------------------------------------------------------------------------
__global__ void k_add_rmsnorm_bf16(const float* __restrict__ hs,
                                   const float* __restrict__ ths,
                                   const float* __restrict__ w,
                                   bf16_t* __restrict__ dst) {
  int row = blockIdx.x;
  int tid = threadIdx.x;
  __shared__ float red[256];
  float t[8];
  float ss = 0.f;
#pragma unroll
  for (int j = 0; j < 8; ++j) {
    int i = j * 256 + tid;
    float v = hs[(size_t)row * Hn + i] + ths[(size_t)row * Hn + i];
    t[j] = v;
    ss += v * v;
  }
  red[tid] = ss; __syncthreads();
  for (int o = 128; o > 0; o >>= 1) {
    if (tid < o) red[tid] += red[tid + o];
    __syncthreads();
  }
  float scale = rsqrtf(red[0] / (float)Hn + EPSf);
#pragma unroll
  for (int j = 0; j < 8; ++j) {
    int i = j * 256 + tid;
    dst[(size_t)row * Hn + i] = (bf16_t)(t[j] * scale * w[i]);
  }
}

// ---------------------------------------------------------------------------
// Generic bf16 WMMA GEMM:  C[M,N] = A[M,K] * Bt[N,K]^T (+ resid)
// block = 256 threads (8 waves along M), wave tile = 64x64 (16 WMMA/k-step).
// grid = (N/64, M/512)
// ---------------------------------------------------------------------------
__global__ void __launch_bounds__(256)
k_gemm_bf16(const bf16_t* __restrict__ A, int lda,
            const bf16_t* __restrict__ Bt, int ldb,
            float* __restrict__ C, int ldc,
            const float* __restrict__ resid, int Kd) {
  int lane = threadIdx.x & 31;
  int wave = threadIdx.x >> 5;
  int n0 = blockIdx.x * 64;
  int m0 = blockIdx.y * 512 + wave * 64;

  v8f acc[4][4] = {};
  for (int k = 0; k < Kd; k += 32) {
    v16bf a[4], b[4];
#pragma unroll
    for (int i = 0; i < 4; ++i) a[i] = load_frag_a(A, lda, m0 + 16 * i, k, lane);
#pragma unroll
    for (int j = 0; j < 4; ++j) b[j] = load_frag_bt(Bt, ldb, n0 + 16 * j, k, lane);
    if (k + 256 < Kd) {  // prefetch ~8 k-steps ahead (global_prefetch_b8)
      __builtin_prefetch(A + (size_t)(m0 + lane) * lda + k + 256, 0, 1);
      __builtin_prefetch(A + (size_t)(m0 + 32 + lane) * lda + k + 256, 0, 1);
      __builtin_prefetch(Bt + (size_t)(n0 + lane) * ldb + k + 256, 0, 1);
      __builtin_prefetch(Bt + (size_t)(n0 + 32 + lane) * ldb + k + 256, 0, 1);
    }
#pragma unroll
    for (int i = 0; i < 4; ++i)
#pragma unroll
      for (int j = 0; j < 4; ++j)
        acc[i][j] = wmma_bf16(a[i], b[j], acc[i][j]);
  }
  int cbase = (lane & 16) ? 8 : 0;
#pragma unroll
  for (int mi = 0; mi < 4; ++mi) {
#pragma unroll
    for (int nj = 0; nj < 4; ++nj) {
      int col = n0 + nj * 16 + (lane & 15);
      int r0  = m0 + mi * 16 + cbase;
#pragma unroll
      for (int r = 0; r < 8; ++r) {
        size_t idx = (size_t)(r0 + r) * ldc + col;
        float v = acc[mi][nj][r];
        if (resid) v += resid[idx];
        C[idx] = v;
      }
    }
  }
}

// ---------------------------------------------------------------------------
// causal depthwise conv1d (K=4) + bias + SiLU on xBC slice of proj
// grid = (ceil(CONVD/256), ROWS)
// ---------------------------------------------------------------------------
__global__ void k_conv_silu(const float* __restrict__ proj,
                            const float* __restrict__ cw,
                            const float* __restrict__ cb,
                            float* __restrict__ xconv) {
  int ch  = blockIdx.x * 256 + threadIdx.x;
  int row = blockIdx.y;
  if (ch >= CONVDn) return;
  int b = row >> 11;          // row / S
  int t = row & (Sn - 1);
  float acc = cb[ch];
#pragma unroll
  for (int j = 0; j < Kn; ++j) {
    int tt = t - (Kn - 1) + j;
    if (tt >= 0)
      acc += cw[ch * Kn + j] * proj[(size_t)(b * Sn + tt) * PROJn + In + ch];
  }
  float s = acc / (1.f + __expf(-acc));
  xconv[(size_t)row * CONVDn + ch] = s;
}

// ---------------------------------------------------------------------------
// prep: dtp = softplus(dt + bias); dtA = dtp * (-exp(A_log));
//       xs = bf16(x * dtp);  Bc/Cc = bf16(B/C)
// grid = ROWS x 256
// ---------------------------------------------------------------------------
__global__ void k_prep(const float* __restrict__ proj,
                       const float* __restrict__ xconv,
                       const float* __restrict__ A_log,
                       const float* __restrict__ dt_bias,
                       float* __restrict__ dtA,
                       bf16_t* __restrict__ xs,
                       bf16_t* __restrict__ Bc,
                       bf16_t* __restrict__ Cc) {
  int row = blockIdx.x;
  int tid = threadIdx.x;
  __shared__ float dtp_s[NHn];
  if (tid < NHn) {
    float x = proj[(size_t)row * PROJn + In + CONVDn + tid] + dt_bias[tid];
    float sp = (x > 20.f) ? x : log1pf(__expf(x));
    dtp_s[tid] = sp;
    dtA[(size_t)row * NHn + tid] = sp * (-__expf(A_log[tid]));
  }
  __syncthreads();
#pragma unroll
  for (int j = 0; j < 16; ++j) {
    int i = j * 256 + tid;
    int h = i >> 6;
    xs[(size_t)row * In + i] = (bf16_t)(xconv[(size_t)row * CONVDn + i] * dtp_s[h]);
  }
  if (tid < Nn) {
    Bc[(size_t)row * Nn + tid] = (bf16_t)xconv[(size_t)row * CONVDn + In + tid];
    Cc[(size_t)row * Nn + tid] = (bf16_t)xconv[(size_t)row * CONVDn + In + Nn + tid];
  }
}

// ---------------------------------------------------------------------------
// per-chunk inclusive cumsum of dtA -> Acs[b][h][c][l]
// grid = B*NH*NC blocks of 256
// ---------------------------------------------------------------------------
__global__ void k_scan_acs(const float* __restrict__ dtA, float* __restrict__ Acs) {
  int bx = blockIdx.x;
  int c = bx & 7, h = (bx >> 3) & 63, b = bx >> 9;
  int tid = threadIdx.x;
  __shared__ float sb[256];
  int row = b * Sn + c * CHUNKn + tid;
  sb[tid] = dtA[(size_t)row * NHn + h];
  __syncthreads();
  for (int o = 1; o < 256; o <<= 1) {
    float t = (tid >= o) ? sb[tid - o] : 0.f;
    __syncthreads();
    sb[tid] += t;
    __syncthreads();
  }
  Acs[((size_t)(b * NHn + h) * NCn + c) * CHUNKn + tid] = sb[tid];
}

// ---------------------------------------------------------------------------
// Chunk kernel A:  Yd (diag block) + per-chunk states.
// One block (256 thr = 8 waves) per (b, c, h).
//   Yd = (tril(exp(acs_l - acs_s)) ∘ (C · B^T)) · x
//   states[p,n] = sum_l x[l,p]*exp(acs_last - acs_l)*B[l,n]
// ---------------------------------------------------------------------------
__global__ void k_chunk_yd_states(const bf16_t* __restrict__ xsg,
                                  const bf16_t* __restrict__ Bcg,
                                  const bf16_t* __restrict__ Ccg,
                                  const float* __restrict__ Acs,
                                  float* __restrict__ yg,
                                  float* __restrict__ states) {
  int bx = blockIdx.x;
  int h = bx & 63, c = (bx >> 6) & 7, b = bx >> 9;
  int tid = threadIdx.x, lane = tid & 31, wave = tid >> 5;
  int base = b * Sn + c * CHUNKn;

  __shared__ __attribute__((aligned(32))) bf16_t xsT[HDn][CHUNKn]; // x^T  32KB
  __shared__ __attribute__((aligned(32))) bf16_t BsT[Nn][CHUNKn];  // B^T  32KB
  __shared__ __attribute__((aligned(32))) bf16_t gst[8][16][32];   // G stage 8KB
  __shared__ float acs_s[CHUNKn];
  __shared__ float ds_s[CHUNKn];

  for (int idx = tid; idx < CHUNKn * HDn; idx += 256) {
    int l = idx >> 6, p = idx & 63;
    xsT[p][l] = xsg[(size_t)(base + l) * In + h * HDn + p];
  }
  for (int idx = tid; idx < CHUNKn * Nn; idx += 256) {
    int l = idx >> 6, n = idx & 63;
    BsT[n][l] = Bcg[(size_t)(base + l) * Nn + n];
  }
  acs_s[tid] = Acs[((size_t)(b * NHn + h) * NCn + c) * CHUNKn + tid];
  __syncthreads();
  ds_s[tid] = __expf(acs_s[CHUNKn - 1] - acs_s[tid]);
  __syncthreads();

  const bf16_t* Crow = Ccg + (size_t)base * Nn;
  const bf16_t* Brow = Bcg + (size_t)base * Nn;

  // ---- Yd ----
  for (int q = 0; q < 2; ++q) {
    int lt = wave * 2 + q;                      // l-tile (16 rows)
    v16bf ca0 = load_frag_a(Crow, Nn, lt * 16, 0,  lane);
    v16bf ca1 = load_frag_a(Crow, Nn, lt * 16, 32, lane);
    v8f acc[4] = {};
    int st2max = (lt * 16 + 15) >> 5;
    for (int st2 = 0; st2 <= st2max; ++st2) {
#pragma unroll
      for (int half = 0; half < 2; ++half) {
        int stt = st2 * 2 + half;
        int scol = stt * 16 + (lane & 15);
        int lrow0 = lt * 16 + ((lane & 16) ? 8 : 0);
        if (stt > lt) {
#pragma unroll
          for (int r = 0; r < 8; ++r)
            gst[wave][(lrow0 - lt * 16) + r][half * 16 + (lane & 15)] = (bf16_t)0.f;
        } else {
          v8f s1 = {};
          int koff = (lane & 16) ? 16 : 0;
          const bf16_t* bp = Brow + (size_t)(stt * 16 + (lane & 15)) * Nn;
          v16bf bb0 = *(const v16bf*)(bp + koff);
          v16bf bb1 = *(const v16bf*)(bp + 32 + koff);
          s1 = wmma_bf16(ca0, bb0, s1);
          s1 = wmma_bf16(ca1, bb1, s1);
#pragma unroll
          for (int r = 0; r < 8; ++r) {
            int l = lrow0 + r;
            float v = (scol <= l) ? s1[r] * __expf(acs_s[l] - acs_s[scol]) : 0.f;
            gst[wave][(l - lt * 16)][half * 16 + (lane & 15)] = (bf16_t)v;
          }
        }
      }
      // A fragment from the masked/decayed score tile
      int arow = lane & 15;
      int aoff = (lane & 16) ? 16 : 0;
      union { v16bf v; v8bf hh[2]; } ga;
      ga.hh[0] = *(const v8bf*)(&gst[wave][arow][aoff]);
      ga.hh[1] = *(const v8bf*)(&gst[wave][arow][8 + aoff]);
      int kb = st2 * 32 + ((lane & 16) ? 16 : 0);
#pragma unroll
      for (int pt = 0; pt < 4; ++pt) {
        v16bf xb = *(const v16bf*)(&xsT[pt * 16 + (lane & 15)][kb]);
        acc[pt] = wmma_bf16(ga.v, xb, acc[pt]);
      }
    }
    // store Yd
#pragma unroll
    for (int pt = 0; pt < 4; ++pt) {
      int col = h * HDn + pt * 16 + (lane & 15);
      int r0  = lt * 16 + ((lane & 16) ? 8 : 0);
#pragma unroll
      for (int r = 0; r < 8; ++r)
        yg[(size_t)(base + r0 + r) * In + col] = acc[pt][r];
    }
  }

  // ---- states = (x * ds)^T @ B ----
  float* sp = states + (size_t)((b * NCn + c) * NHn + h) * HDn * Nn;
  for (int tt = wave; tt < 16; tt += 8) {
    int mt = tt >> 2, nt = tt & 3;
    v8f acc = {};
    for (int kb = 0; kb < CHUNKn; kb += 32) {
      int p = mt * 16 + (lane & 15);
      int off = (lane & 16) ? 16 : 0;
      union { v16bf v; v8bf hh[2]; } ua;
      ua.hh[0] = *(const v8bf*)(&xsT[p][kb + off]);
      ua.hh[1] = *(const v8bf*)(&xsT[p][kb + 8 + off]);
#pragma unroll
      for (int e = 0; e < 8; ++e)
        ua.v[e] = (bf16_t)((float)ua.v[e] * ds_s[kb + off + e]);
#pragma unroll
      for (int e = 0; e < 8; ++e)
        ua.v[8 + e] = (bf16_t)((float)ua.v[8 + e] * ds_s[kb + 8 + off + e]);
      int koff = kb + ((lane & 16) ? 16 : 0);
      v16bf bb = *(const v16bf*)(&BsT[nt * 16 + (lane & 15)][koff]);
      acc = wmma_bf16(ua.v, bb, acc);
    }
    int cn = nt * 16 + (lane & 15);
    int r0 = mt * 16 + ((lane & 16) ? 8 : 0);
#pragma unroll
    for (int r = 0; r < 8; ++r) sp[(r0 + r) * Nn + cn] = acc[r];
  }
}

// ---------------------------------------------------------------------------
// inter-chunk state recurrence:  P_0 = 0;  P_{c+1} = exp(Asum_c) P_c + states_c
// grid = B*NH blocks of 256; each thread carries 16 of the 64x64 state elems.
// ---------------------------------------------------------------------------
__global__ void k_prev_scan(const float* __restrict__ states,
                            const float* __restrict__ Acs,
                            bf16_t* __restrict__ prevb) {
  int bx = blockIdx.x;
  int b = bx >> 6, h = bx & 63;
  int tid = threadIdx.x;
  float P[16];
#pragma unroll
  for (int e = 0; e < 16; ++e) P[e] = 0.f;
  for (int c = 0; c < NCn; ++c) {
    size_t off = (size_t)((b * NCn + c) * NHn + h) * HDn * Nn;
    bf16_t* pb = prevb + off;
    const float* st = states + off;
    float dec = __expf(Acs[((size_t)(b * NHn + h) * NCn + c) * CHUNKn + (CHUNKn - 1)]);
#pragma unroll
    for (int e = 0; e < 16; ++e) pb[tid * 16 + e] = (bf16_t)P[e];
#pragma unroll
    for (int e = 0; e < 16; ++e) P[e] = P[e] * dec + st[tid * 16 + e];
  }
}

// ---------------------------------------------------------------------------
// Chunk kernel B:  y += exp(acs_l) * (C @ prev^T) + D[h] * x_raw
// ---------------------------------------------------------------------------
__global__ void k_chunk_yo(const bf16_t* __restrict__ Ccg,
                           const bf16_t* __restrict__ prevb,
                           const float* __restrict__ Acs,
                           const float* __restrict__ xconv,
                           const float* __restrict__ Dv,
                           float* __restrict__ yg) {
  int bx = blockIdx.x;
  int h = bx & 63, c = (bx >> 6) & 7, b = bx >> 9;
  int tid = threadIdx.x, lane = tid & 31, wave = tid >> 5;
  int base = b * Sn + c * CHUNKn;
  __shared__ float acs_s[CHUNKn];
  acs_s[tid] = Acs[((size_t)(b * NHn + h) * NCn + c) * CHUNKn + tid];
  __syncthreads();

  const bf16_t* Crow = Ccg + (size_t)base * Nn;
  const bf16_t* pb = prevb + (size_t)((b * NCn + c) * NHn + h) * HDn * Nn;
  float Dh = Dv[h];

  for (int q = 0; q < 2; ++q) {
    int lt = wave * 2 + q;
    v16bf ca0 = load_frag_a(Crow, Nn, lt * 16, 0,  lane);
    v16bf ca1 = load_frag_a(Crow, Nn, lt * 16, 32, lane);
    v8f acc[4] = {};
#pragma unroll
    for (int pt = 0; pt < 4; ++pt) {
      int prow = pt * 16 + (lane & 15);
      int koff = (lane & 16) ? 16 : 0;
      v16bf b0 = *(const v16bf*)(pb + (size_t)prow * Nn + koff);
      v16bf b1 = *(const v16bf*)(pb + (size_t)prow * Nn + 32 + koff);
      acc[pt] = wmma_bf16(ca0, b0, acc[pt]);
      acc[pt] = wmma_bf16(ca1, b1, acc[pt]);
    }
#pragma unroll
    for (int pt = 0; pt < 4; ++pt) {
      int pcol = pt * 16 + (lane & 15);
      int r0 = lt * 16 + ((lane & 16) ? 8 : 0);
#pragma unroll
      for (int r = 0; r < 8; ++r) {
        int l = r0 + r;
        size_t yi = (size_t)(base + l) * In + h * HDn + pcol;
        float yo = acc[pt][r] * __expf(acs_s[l]);
        float xr = xconv[(size_t)(base + l) * CONVDn + h * HDn + pcol];
        yg[yi] += yo + Dh * xr;
      }
    }
  }
}

// ---------------------------------------------------------------------------
// gate + rmsnorm:  ygb = bf16( rmsnorm(y * silu(z)) * norm_w ),  row = 4096
// ---------------------------------------------------------------------------
__global__ void k_gate_norm(const float* __restrict__ yg,
                            const float* __restrict__ proj,
                            const float* __restrict__ w,
                            bf16_t* __restrict__ ygb) {
  int row = blockIdx.x;
  int tid = threadIdx.x;
  __shared__ float red[256];
  float t[16];
  float ss = 0.f;
#pragma unroll
  for (int j = 0; j < 16; ++j) {
    int i = j * 256 + tid;
    float z = proj[(size_t)row * PROJn + i];
    float v = yg[(size_t)row * In + i] * (z / (1.f + __expf(-z)));
    t[j] = v;
    ss += v * v;
  }
  red[tid] = ss; __syncthreads();
  for (int o = 128; o > 0; o >>= 1) {
    if (tid < o) red[tid] += red[tid + o];
    __syncthreads();
  }
  float scale = rsqrtf(red[0] / (float)In + EPSf);
#pragma unroll
  for (int j = 0; j < 16; ++j) {
    int i = j * 256 + tid;
    ygb[(size_t)row * In + i] = (bf16_t)(t[j] * scale * w[i]);
  }
}

// ---------------------------------------------------------------------------
// host-side orchestration
// ---------------------------------------------------------------------------
static inline size_t align256(size_t x) { return (x + 255) & ~(size_t)255; }

extern "C" void kernel_launch(void* const* d_in, const int* in_sizes, int n_in,
                              void* d_out, int out_size, void* d_ws, size_t ws_size,
                              hipStream_t stream) {
  (void)in_sizes; (void)n_in; (void)out_size; (void)ws_size;
  const float* hidden   = (const float*)d_in[0];
  const float* thidden  = (const float*)d_in[1];
  const float* rms_w    = (const float*)d_in[2];
  const float* in_w     = (const float*)d_in[3];
  const float* conv_w   = (const float*)d_in[4];
  const float* conv_b   = (const float*)d_in[5];
  const float* A_log    = (const float*)d_in[6];
  const float* Dv       = (const float*)d_in[7];
  const float* dt_bias  = (const float*)d_in[8];
  const float* norm_w   = (const float*)d_in[9];
  const float* out_w    = (const float*)d_in[10];
  float* out = (float*)d_out;

  char* ws = (char*)d_ws;
  size_t off = 0;
  auto alloc = [&](size_t bytes) { size_t o = off; off = align256(off + bytes); return o; };

  bf16_t* winbf  = (bf16_t*)(ws + alloc((size_t)PROJn * Hn * 2));
  bf16_t* outwbf = (bf16_t*)(ws + alloc((size_t)Hn * In * 2));
  bf16_t* hsn    = (bf16_t*)(ws + alloc((size_t)ROWSn * Hn * 2));
  float*  proj   = (float*) (ws + alloc((size_t)ROWSn * PROJn * 4));
  float*  xconv  = (float*) (ws + alloc((size_t)ROWSn * CONVDn * 4));
  float*  dtA    = (float*) (ws + alloc((size_t)ROWSn * NHn * 4));
  bf16_t* xs     = (bf16_t*)(ws + alloc((size_t)ROWSn * In * 2));
  bf16_t* Bc     = (bf16_t*)(ws + alloc((size_t)ROWSn * Nn * 2));
  bf16_t* Cc     = (bf16_t*)(ws + alloc((size_t)ROWSn * Nn * 2));
  float*  Acs    = (float*) (ws + alloc((size_t)Bn * NHn * NCn * CHUNKn * 4));
  float*  states = (float*) (ws + alloc((size_t)Bn * NCn * NHn * HDn * Nn * 4));
  bf16_t* prevb  = (bf16_t*)(ws + alloc((size_t)Bn * NCn * NHn * HDn * Nn * 2));
  float*  yg     = (float*) (ws + alloc((size_t)ROWSn * In * 4));
  bf16_t* ygb    = (bf16_t*)(ws + alloc((size_t)ROWSn * In * 2));

  // weights -> bf16
  k_f32_to_bf16<<<2048, 256, 0, stream>>>(in_w,  winbf,  (size_t)PROJn * Hn);
  k_f32_to_bf16<<<2048, 256, 0, stream>>>(out_w, outwbf, (size_t)Hn * In);

  // 1. add + rmsnorm -> bf16
  k_add_rmsnorm_bf16<<<ROWSn, 256, 0, stream>>>(hidden, thidden, rms_w, hsn);

  // 2. proj = hsn @ in_w^T   (M=4096, N=8384, K=2048)
  k_gemm_bf16<<<dim3(PROJn / 64, ROWSn / 512), 256, 0, stream>>>(
      hsn, Hn, winbf, Hn, proj, PROJn, nullptr, Hn);

  // 3. depthwise conv + SiLU
  k_conv_silu<<<dim3((CONVDn + 255) / 256, ROWSn), 256, 0, stream>>>(
      proj, conv_w, conv_b, xconv);

  // 4. dt/softplus, x*dt, B/C bf16
  k_prep<<<ROWSn, 256, 0, stream>>>(proj, xconv, A_log, dt_bias, dtA, xs, Bc, Cc);

  // 5. per-chunk cumsum
  k_scan_acs<<<Bn * NHn * NCn, 256, 0, stream>>>(dtA, Acs);

  // 6. diag-block output + chunk states (WMMA-heavy)
  k_chunk_yd_states<<<Bn * NCn * NHn, 256, 0, stream>>>(xs, Bc, Cc, Acs, yg, states);

  // 7. inter-chunk recurrence
  k_prev_scan<<<Bn * NHn, 256, 0, stream>>>(states, Acs, prevb);

  // 8. off-diag contribution + D*x
  k_chunk_yo<<<Bn * NCn * NHn, 256, 0, stream>>>(Cc, prevb, Acs, xconv, Dv, yg);

  // 9. gate + rmsnorm -> bf16
  k_gate_norm<<<ROWSn, 256, 0, stream>>>(yg, proj, norm_w, ygb);

  // 10. out = ygb @ out_w^T + residual   (M=4096, N=2048, K=4096)
  k_gemm_bf16<<<dim3(Hn / 64, ROWSn / 512), 256, 0, stream>>>(
      ygb, In, outwbf, In, out, Hn, hidden, In);
}
